// NeuralMemory_62594853372106
// MI455X (gfx1250) — matrix operand order repaired and checked by reading, not compiled, not run on hardware
//
#include <hip/hip_runtime.h>
#include <hip/hip_bf16.h>

typedef __attribute__((ext_vector_type(16))) __bf16 v16bf;
typedef __attribute__((ext_vector_type(8)))  __bf16 v8bf;
typedef __attribute__((ext_vector_type(8)))  float  v8f;

#define B_ROWS 4096
#define DIM    512
#define NKEYS  65536
#define TOPK   32
#define KT     (DIM / 32)        // 16 K-tiles of 32
#define WAVES  8
#define STRIPE (NKEYS / WAVES)   // 8192 keys per wave
#define NTILES (STRIPE / 16)     // 512 N-tiles per wave
#define BDEPTH 8                 // B-fragment pipeline depth (must divide KT)
#define NEG_INF (-3.4e38f)

// ---------------------------------------------------------------------------
// Kernel 1: L2-normalize one row of f32 and emit bf16 (RNE via hw convert).
// ---------------------------------------------------------------------------
__global__ void l2norm_bf16_kernel(const float* __restrict__ src,
                                   __bf16* __restrict__ dst, int ncols) {
    const int row = blockIdx.x;
    const float* x = src + (size_t)row * ncols;
    __shared__ float red[256];
    float s = 0.f;
    for (int d = threadIdx.x; d < ncols; d += blockDim.x) {
        float v = x[d];
        s += v * v;
    }
    red[threadIdx.x] = s;
    __syncthreads();
    for (int off = 128; off > 0; off >>= 1) {
        if (threadIdx.x < off) red[threadIdx.x] += red[threadIdx.x + off];
        __syncthreads();
    }
    const float inv = 1.0f / fmaxf(sqrtf(red[0]), 1e-12f);
    __bf16* y = dst + (size_t)row * ncols;
    for (int d = threadIdx.x; d < ncols; d += blockDim.x)
        y[d] = (__bf16)(x[d] * inv);
}

// ---------------------------------------------------------------------------
// Kernel 2: fused  scores-GEMM (bf16 WMMA) -> top-32 -> softmax -> gather.
// One workgroup (8 wave32s) per 16 query rows; each wave streams an 8192-key
// stripe of the (L2-resident) bf16 key matrix with a depth-8 software
// pipeline on the B fragments so WMMAs overlap global loads.
// ---------------------------------------------------------------------------
__launch_bounds__(256, 1)
__global__ void attn_topk_kernel(const __bf16* __restrict__ qn,
                                 const __bf16* __restrict__ kn,
                                 const float*  __restrict__ vals,
                                 float* __restrict__ outV,
                                 int*   __restrict__ outIdx,
                                 float* __restrict__ outAtt) {
    __shared__ float sTile[WAVES][16][16];     //  8 KB score staging
    __shared__ float sTopS[WAVES][16][TOPK];   // 16 KB per-wave top scores
    __shared__ int   sTopI[WAVES][16][TOPK];   // 16 KB per-wave top indices
    __shared__ float sAtt[16][TOPK];           //  2 KB merged softmax weights
    __shared__ int   sIdx[16][TOPK];           //  2 KB merged indices

    const int tid  = threadIdx.x;
    const int wave = tid >> 5;
    const int lane = tid & 31;
    const int m0   = blockIdx.x * 16;          // first query row of this block

    // init top-k lists
    for (int i = tid; i < WAVES * 16 * TOPK; i += 256) {
        ((float*)sTopS)[i] = NEG_INF;
        ((int*)sTopI)[i]   = 0;
    }
    __syncthreads();

    // ---- A fragments: 16 rows x 512 (bf16), contiguous k order (matches B) -----------
    // lane L: M = L&15, half = L>>4.  element s: K = half*16 + s
    const int  mrow  = m0 + (lane & 15);
    const int  halfA = lane >> 4;
    v16bf afrag[KT];
#pragma unroll
    for (int t = 0; t < KT; ++t) {
        const __bf16* base = qn + (size_t)mrow * DIM + t * 32 + halfA * 16;
        afrag[t] = *(const v16bf*)(base);      // K = t*32 + half*16 .. +15
    }

    // per-lane running min of the top-32 list (lanes 0..15 own row = lane)
    float minV = NEG_INF;
    int   minP = 0;

    const int khalf   = lane >> 4;   // B layout: lanes 16-31 carry K=16..31
    const int ncol    = lane & 15;   // B layout: N = lane & 15
    const int stripe0 = wave * STRIPE;

    // Per-lane B base: key row (stripe0+ncol), K-offset khalf*16.
    // Fragment (nt, t) lives at lanePtr + nt*16*DIM + t*32  (32B contiguous).
    const __bf16* lanePtr = kn + (size_t)(stripe0 + ncol) * DIM + khalf * 16;

    // ---- prime the depth-8 B pipeline (tile 0, fragments 0..7) ------------
    v16bf bbuf[BDEPTH];
#pragma unroll
    for (int i = 0; i < BDEPTH; ++i)
        bbuf[i] = *(const v16bf*)(lanePtr + i * 32);

    for (int nt = 0; nt < NTILES; ++nt) {
        const int nb = stripe0 + nt * 16;
        const __bf16* brow0 = lanePtr + (size_t)nt * 16 * DIM;
        // next-tile base (clamped at stripe end: re-reads tile nt, harmless)
        const __bf16* brow1 = (nt + 1 < NTILES) ? (brow0 + 16 * DIM) : brow0;
        __builtin_prefetch((const void*)(brow1 + 16 * DIM), 0, 3);

        v8f c = {};
#pragma unroll
        for (int t = 0; t < KT; ++t) {
            // issue load for fragment t+BDEPTH (wraps into next tile's 0..7)
            v16bf bn;
            if (t < KT - BDEPTH)
                bn = *(const v16bf*)(brow0 + (t + BDEPTH) * 32);
            else
                bn = *(const v16bf*)(brow1 + (t + BDEPTH - KT) * 32);
            // consume fragment t
            c = __builtin_amdgcn_wmma_f32_16x16x32_bf16(
                    false, afrag[t], false, bbuf[t % BDEPTH],
                    (short)0, c, false, false);
            bbuf[t % BDEPTH] = bn;
        }

        // stage 16x16 f32 scores: VGPR r -> (M = r + khalf*8, N = ncol)
        const int rowOff = khalf * 8;
#pragma unroll
        for (int r = 0; r < 8; ++r)
            sTile[wave][r + rowOff][ncol] = c[r];
        asm volatile("s_wait_dscnt 0" ::: "memory");
        __builtin_amdgcn_wave_barrier();

        // lanes 0..15: update running top-32 for their row
        if (lane < 16) {
            float* ts = sTopS[wave][lane];
            int*   ti = sTopI[wave][lane];
            for (int j = 0; j < 16; ++j) {
                float s = sTile[wave][lane][j];
                if (s > minV) {
                    ts[minP] = s;
                    ti[minP] = nb + j;
                    float mv = ts[0]; int mp = 0;
#pragma unroll
                    for (int q2 = 1; q2 < TOPK; ++q2) {
                        float v = ts[q2];
                        if (v < mv) { mv = v; mp = q2; }
                    }
                    minV = mv; minP = mp;
                }
            }
        }
        __builtin_amdgcn_wave_barrier();
    }
    __syncthreads();

    // ---- merge 8 per-wave lists -> global top-32 (descending), softmax ----
    if (tid < 16) {
        const int r = tid;
        float att[TOPK];
        int   idxs[TOPK];
        for (int j = 0; j < TOPK; ++j) {
            float best = NEG_INF; int bw = 0, bk = 0;
            for (int w = 0; w < WAVES; ++w)
                for (int k2 = 0; k2 < TOPK; ++k2) {
                    float v = sTopS[w][r][k2];
                    if (v > best) { best = v; bw = w; bk = k2; }
                }
            att[j]  = best;
            idxs[j] = sTopI[bw][r][bk];
            sTopS[bw][r][bk] = NEG_INF;
        }
        const float mx = att[0];          // sorted descending
        float sum = 0.f;
        for (int j = 0; j < TOPK; ++j) {
            float e = __expf(att[j] - mx);
            att[j] = e; sum += e;
        }
        const float inv = 1.0f / sum;
        for (int j = 0; j < TOPK; ++j) {
            float a = att[j] * inv;
            sAtt[r][j] = a;
            sIdx[r][j] = idxs[j];
            outAtt[(size_t)(m0 + r) * TOPK + j] = a;
            outIdx[(size_t)(m0 + r) * TOPK + j] = idxs[j];
        }
    }
    __syncthreads();

    // ---- weighted gather of vals: 256 threads x float2 covers D=512 ------
    const int d0 = tid * 2;
    for (int m = 0; m < 16; ++m) {
        float acc0 = 0.f, acc1 = 0.f;
        for (int k2 = 0; k2 < TOPK; ++k2) {
            const float  w  = sAtt[m][k2];
            const float2 vv = *(const float2*)(vals + (size_t)sIdx[m][k2] * DIM + d0);
            acc0 += w * vv.x;
            acc1 += w * vv.y;
        }
        float* op = outV + (size_t)(m0 + m) * DIM + d0;
        op[0] = acc0;
        op[1] = acc1;
    }
}

// ---------------------------------------------------------------------------
extern "C" void kernel_launch(void* const* d_in, const int* in_sizes, int n_in,
                              void* d_out, int out_size, void* d_ws, size_t ws_size,
                              hipStream_t stream) {
    const float* q    = (const float*)d_in[0];   // [4096, 512]
    const float* keys = (const float*)d_in[1];   // [65536, 512]
    const float* vals = (const float*)d_in[2];   // [65536, 512]
    // d_in[3] = topk (fixed 32)

    float* out    = (float*)d_out;
    float* outV   = out;                                             // 4096*512
    int*   outIdx = (int*)(out + (size_t)B_ROWS * DIM);              // 4096*32 (int32 bits)
    float* outAtt = out + (size_t)B_ROWS * DIM + (size_t)B_ROWS * TOPK; // 4096*32

    __bf16* qn = (__bf16*)d_ws;                    //  4 MB
    __bf16* kn = qn + (size_t)B_ROWS * DIM;        // 64 MB (L2-resident)

    l2norm_bf16_kernel<<<B_ROWS, 256, 0, stream>>>(q, qn, DIM);
    l2norm_bf16_kernel<<<NKEYS, 256, 0, stream>>>(keys, kn, DIM);
    attn_topk_kernel<<<B_ROWS / 16, 256, 0, stream>>>(qn, kn, vals,
                                                      outV, outIdx, outAtt);
}
